// GCN2Layer_12652973654218
// MI455X (gfx1250) — matrix operand order repaired and checked by reading, not compiled, not run on hardware
//
#include <hip/hip_runtime.h>
#include <hip/hip_bf16.h>
#include <math.h>

// ---------------------------------------------------------------------------
// 2-layer GCN, reordered for bandwidth:
//   deg -> dinv -> agg1 = Â·x (2-wide edge pass) -> h = relu(agg1@W1+b1) [f16]
//   t = h@W2 (WMMA f32_16x16x32_f16) -> out2 = Â·t (2-wide edge pass) + b2
//   -> segment-max pool (ordered-uint atomicMax) -> log_softmax
// W2's WMMA-B operand image is precomputed per-lane (branch-free WMMA kernel).
// ---------------------------------------------------------------------------

typedef __attribute__((ext_vector_type(16))) _Float16 v16h;
typedef __attribute__((ext_vector_type(8)))  float    v8f;

union V16H { v16h v; _Float16 e[16]; };
union V8F  { v8f  v; float    e[8];  };

// order-preserving float <-> uint encoding for atomicMax on floats
static __device__ __forceinline__ unsigned fenc(float f) {
  unsigned u = __float_as_uint(f);
  return (u & 0x80000000u) ? ~u : (u | 0x80000000u);
}
static __device__ __forceinline__ float fdec(unsigned u) {
  unsigned b = (u & 0x80000000u) ? (u & 0x7FFFFFFFu) : ~u;
  return __uint_as_float(b);
}

// deg := 1.0 (self loop), pooled := 0 (encoded minimum)
__global__ __launch_bounds__(256) void gcn_init(float* deg, unsigned* pooled,
                                                int n, int p2) {
  int i = blockIdx.x * blockDim.x + threadIdx.x;
  if (i < n)  deg[i] = 1.0f;
  if (i < p2) pooled[i] = 0u;
}

__global__ __launch_bounds__(256) void gcn_deg(const int* __restrict__ col,
                                               float* deg, int e) {
  int i = blockIdx.x * blockDim.x + threadIdx.x;
  if (i >= e) return;
  atomicAdd(deg + col[i], 1.0f);
}

// dinv = deg^-1/2 (in place); seed agg1 with self-loop term dinv^2 * x
__global__ __launch_bounds__(256) void gcn_dinv_selfloop(
    const float* __restrict__ x, float* dinv, float* __restrict__ agg1, int n) {
  int i = blockIdx.x * blockDim.x + threadIdx.x;
  if (i >= n) return;
  float d = rsqrtf(dinv[i]);   // deg >= 1 always (self loop)
  dinv[i] = d;
  float w = d * d;
  agg1[(size_t)i * 2 + 0] = w * x[(size_t)i * 2 + 0];
  agg1[(size_t)i * 2 + 1] = w * x[(size_t)i * 2 + 1];
}

// dst2[c] += dinv[r]*dinv[c] * src2[r]   (2-wide messages, f32 atomics)
__global__ __launch_bounds__(256) void gcn_edge_agg(
    const int* __restrict__ row, const int* __restrict__ col,
    const float* __restrict__ dinv, const float* __restrict__ src2,
    float* __restrict__ dst2, int e) {
  int i = blockIdx.x * blockDim.x + threadIdx.x;
  if (i >= e) return;
  int r = row[i], c = col[i];
  float w = dinv[r] * dinv[c];
  float2 s = *(const float2*)(src2 + (size_t)r * 2);
  atomicAdd(dst2 + (size_t)c * 2 + 0, w * s.x);
  atomicAdd(dst2 + (size_t)c * 2 + 1, w * s.y);
}

// h[i,0:64] = relu(agg1[i,:] @ W1 + b1), stored f16; 8 threads/node, B128 stores
__global__ __launch_bounds__(256) void gcn_hidden(
    const float* __restrict__ agg1, const float* __restrict__ W1,
    const float* __restrict__ b1, _Float16* __restrict__ h, int n) {
  int tid = blockIdx.x * blockDim.x + threadIdx.x;
  int i = tid >> 3;
  if (i >= n) return;
  int jb = (tid & 7) * 8;
  float a0 = agg1[(size_t)i * 2 + 0];
  float a1 = agg1[(size_t)i * 2 + 1];
  union { uint4 q; _Float16 e[8]; } o;
#pragma unroll
  for (int jj = 0; jj < 8; ++jj) {
    int j = jb + jj;
    float v = fmaf(a0, W1[j], fmaf(a1, W1[64 + j], b1[j]));
    o.e[jj] = (_Float16)(v > 0.0f ? v : 0.0f);
  }
  *(uint4*)(h + (size_t)i * 64 + jb) = o.q;
}

// Precompute per-lane WMMA B operand image for W2 (64x2 padded to 64x16, f16).
// ISA B 32x16 layout: lane = column N (lane%16), halves p = K rows;
// lanes0-15 -> K 0..7,16..23, lanes16-31 -> K 8..15,24..31. Bpre[lane][0:16]  = K 0..31
// chunk for first WMMA, Bpre[lane][16:32] = K 32..63 chunk for second WMMA.
__global__ void gcn_prep_B(const float* __restrict__ W2,
                           _Float16* __restrict__ Bpre) {
  int lane = threadIdx.x;
  if (lane >= 32) return;
  const int  ncol = lane & 15;
  const int  kb   = (lane >= 16) ? 8 : 0;
#pragma unroll
  for (int p = 0; p < 16; ++p) {
    float w0 = 0.0f, w1 = 0.0f;
    if (ncol < 2) {
      const int kr = kb + p + (p >= 8 ? 8 : 0);
      w0 = W2[kr * 2 + ncol];
      w1 = W2[(kr + 32) * 2 + ncol];
    }
    Bpre[lane * 32 + p]      = (_Float16)w0;
    Bpre[lane * 32 + 16 + p] = (_Float16)w1;
  }
}

// t[16-tile,0:2] = h[16x64] @ W2[64x2] via two chained v_wmma_f32_16x16x32_f16.
__global__ __launch_bounds__(256) void gcn_wmma_hW2(
    const _Float16* __restrict__ h, const _Float16* __restrict__ Bpre,
    float* __restrict__ t, int n, int ntiles) {
  const int wave = threadIdx.x >> 5;
  const int lane = threadIdx.x & 31;
  const int tile = blockIdx.x * (blockDim.x >> 5) + wave;
  if (tile >= ntiles) return;            // wave-uniform: EXEC all-1 for WMMA

  const bool hiHalf = lane >= 16;
  const int  m      = lane & 15;
  int row = tile * 16 + m;
  if (row >= n) row = n - 1;             // clamp tail reads; stores guarded
  const _Float16* hrow = h + (size_t)row * 64;

  // ISA 16-bit A 16x32 layout: lanes0-15 halves0-7 = K0..7, halves8-15 = K16..23
  //                            lanes16-31 halves0-7 = K8..15, halves8-15 = K24..31
  const int klo = hiHalf ? 8 : 0;
  const int khi = hiHalf ? 24 : 16;

  V16H A0, A1, B0, B1;
  *(uint4*)(&A0.e[0]) = *(const uint4*)(hrow + klo);
  *(uint4*)(&A0.e[8]) = *(const uint4*)(hrow + khi);
  *(uint4*)(&A1.e[0]) = *(const uint4*)(hrow + 32 + klo);
  *(uint4*)(&A1.e[8]) = *(const uint4*)(hrow + 32 + khi);

  const _Float16* bl = Bpre + lane * 32;
  *(uint4*)(&B0.e[0]) = *(const uint4*)(bl + 0);
  *(uint4*)(&B0.e[8]) = *(const uint4*)(bl + 8);
  *(uint4*)(&B1.e[0]) = *(const uint4*)(bl + 16);
  *(uint4*)(&B1.e[8]) = *(const uint4*)(bl + 24);

  v8f c = {};
  c = __builtin_amdgcn_wmma_f32_16x16x32_f16(false, A0.v, false, B0.v,
                                             (short)0, c, false, false);
  c = __builtin_amdgcn_wmma_f32_16x16x32_f16(false, A1.v, false, B1.v,
                                             (short)0, c, false, false);

  // C/D layout: VGPR j, lanes0-15 -> M=j, lanes16-31 -> M=j+8; N = lane%16
  const bool full = (tile * 16 + 16) <= n;   // wave-uniform fast path
  if (m < 2) {
    V8F C; C.v = c;
    const int rbase = tile * 16 + (hiHalf ? 8 : 0);
    float* tp = t + (size_t)rbase * 2 + m;
    if (full) {
#pragma unroll
      for (int j = 0; j < 8; ++j) tp[j * 2] = C.e[j];
    } else {
#pragma unroll
      for (int j = 0; j < 8; ++j)
        if (rbase + j < n) tp[j * 2] = C.e[j];
    }
  }
}

// out2 seed with self-loop term dinv^2 * t
__global__ __launch_bounds__(256) void gcn_selfloop2(
    const float* __restrict__ t, const float* __restrict__ dinv,
    float* __restrict__ out2, int n) {
  int i = blockIdx.x * blockDim.x + threadIdx.x;
  if (i >= n) return;
  float w = dinv[i] * dinv[i];
  out2[(size_t)i * 2 + 0] = w * t[(size_t)i * 2 + 0];
  out2[(size_t)i * 2 + 1] = w * t[(size_t)i * 2 + 1];
}

// global max-pool per graph via ordered-uint atomicMax (bias b2 added here)
__global__ __launch_bounds__(256) void gcn_pool(
    const float* __restrict__ out2, const float* __restrict__ b2,
    const int* __restrict__ batch, unsigned* pooled, int n) {
  int i = blockIdx.x * blockDim.x + threadIdx.x;
  if (i >= n) return;
  int g = batch[i];
  atomicMax(pooled + (size_t)g * 2 + 0, fenc(out2[(size_t)i * 2 + 0] + b2[0]));
  atomicMax(pooled + (size_t)g * 2 + 1, fenc(out2[(size_t)i * 2 + 1] + b2[1]));
}

__global__ __launch_bounds__(256) void gcn_logsoftmax(
    const unsigned* __restrict__ pooled, float* __restrict__ out, int g) {
  int i = blockIdx.x * blockDim.x + threadIdx.x;
  if (i >= g) return;
  float a0 = fdec(pooled[i * 2 + 0]);
  float a1 = fdec(pooled[i * 2 + 1]);
  float mx = fmaxf(a0, a1);
  float lse = mx + logf(expf(a0 - mx) + expf(a1 - mx));
  out[i * 2 + 0] = a0 - lse;
  out[i * 2 + 1] = a1 - lse;
}

extern "C" void kernel_launch(void* const* d_in, const int* in_sizes, int n_in,
                              void* d_out, int out_size, void* d_ws, size_t ws_size,
                              hipStream_t stream) {
  (void)n_in; (void)ws_size;
  const float* x     = (const float*)d_in[0];   // [n,2]
  const int*   ei    = (const int*)d_in[1];     // [2,e]
  const int*   batch = (const int*)d_in[2];     // [n]
  const float* W1    = (const float*)d_in[3];   // [2,64]
  const float* b1    = (const float*)d_in[4];   // [64]
  const float* W2    = (const float*)d_in[5];   // [64,2]
  const float* b2    = (const float*)d_in[6];   // [2]
  float*       out   = (float*)d_out;           // [g,2]

  const int n = in_sizes[0] / 2;
  const int e = in_sizes[1] / 2;
  const int g = out_size / 2;
  const int* row = ei;
  const int* col = ei + e;

  auto al = [](size_t v) { return (v + 255) & ~(size_t)255; };
  char* p = (char*)d_ws;
  float*    dinv   = (float*)p;     p += al((size_t)n * 4);    // deg -> dinv
  float*    agg1   = (float*)p;     p += al((size_t)n * 8);    // Â·x  [n,2]
  _Float16* hbuf   = (_Float16*)p;  p += al((size_t)n * 128);  // h    [n,64] f16
  float*    tbuf   = (float*)p;     p += al((size_t)n * 8);    // h@W2 [n,2]
  float*    out2   = (float*)p;     p += al((size_t)n * 8);    // Â·t  [n,2]
  unsigned* pooled = (unsigned*)p;  p += al((size_t)g * 8);    // [g,2] encoded
  _Float16* Bpre   = (_Float16*)p;                             // [32][32] f16

  const int B = 256;
  const int initN  = (n > 2 * g) ? n : 2 * g;
  const int ntiles = (n + 15) / 16;

  gcn_init<<<(initN + B - 1) / B, B, 0, stream>>>(dinv, pooled, n, 2 * g);
  gcn_prep_B<<<1, 32, 0, stream>>>(W2, Bpre);
  gcn_deg<<<(e + B - 1) / B, B, 0, stream>>>(col, dinv, e);
  gcn_dinv_selfloop<<<(n + B - 1) / B, B, 0, stream>>>(x, dinv, agg1, n);
  gcn_edge_agg<<<(e + B - 1) / B, B, 0, stream>>>(row, col, dinv, x, agg1, e);
  gcn_hidden<<<(n * 8 + B - 1) / B, B, 0, stream>>>(agg1, W1, b1, hbuf, n);
  gcn_wmma_hW2<<<(ntiles + 7) / 8, B, 0, stream>>>(hbuf, Bpre, tbuf, n, ntiles);
  gcn_selfloop2<<<(n + B - 1) / B, B, 0, stream>>>(tbuf, dinv, out2, n);
  gcn_edge_agg<<<(e + B - 1) / B, B, 0, stream>>>(row, col, dinv, tbuf, out2, e);
  gcn_pool<<<(n + B - 1) / B, B, 0, stream>>>(out2, b2, batch, pooled, n);
  gcn_logsoftmax<<<(g + B - 1) / B, B, 0, stream>>>(pooled, out, g);
}
